// NeurosynapticEventEncoder_86535001080334
// MI455X (gfx1250) — matrix operand order repaired and checked by reading, not compiled, not run on hardware
//
#include <hip/hip_runtime.h>
#include <hip/hip_bf16.h>

// Sizes from the reference
#define S     512
#define DIN   256
#define H     512

typedef __attribute__((ext_vector_type(16))) _Float16 v16h;
typedef __attribute__((ext_vector_type(8)))  _Float16 v8h;
typedef __attribute__((ext_vector_type(8)))  float    v8f;
typedef __attribute__((ext_vector_type(4)))  float    v4f;

// ---------------------------------------------------------------------------
// 1) Stable rank-sort (argsort of 512 timestamps), O(S^2) in one block.
// ---------------------------------------------------------------------------
__global__ __launch_bounds__(512)
void sort_rank_kernel(const float* __restrict__ ts, const float* __restrict__ amps,
                      int* __restrict__ order, float* __restrict__ ts_s,
                      float* __restrict__ amps_s)
{
    __shared__ float sh[S];
    int i = threadIdx.x;
    float v = ts[i];
    sh[i] = v;
    __syncthreads();
    int rank = 0;
    for (int j = 0; j < S; ++j) {
        float u = sh[j];
        rank += (u < v) || (u == v && j < i);   // stable tiebreak == jnp.argsort
    }
    order[rank]  = i;
    ts_s[rank]   = v;
    amps_s[rank] = amps[i];
}

// ---------------------------------------------------------------------------
// 2) Gather sorted feature rows, converting to f16 for WMMA consumption.
// ---------------------------------------------------------------------------
__global__ __launch_bounds__(256)
void gather_feats_kernel(const float* __restrict__ feats, const int* __restrict__ order,
                         _Float16* __restrict__ out)
{
    int r = blockIdx.x;          // sorted row
    int src = order[r];
    int c = threadIdx.x;         // DIN == 256
    out[r * DIN + c] = (_Float16)feats[src * DIN + c];
}

// ---------------------------------------------------------------------------
// Transpose + convert: in[K,N] f32 -> out[N,K] f16.  LDS-tiled, coalesced.
// Grids: (N/32, K/32), block 256.
// ---------------------------------------------------------------------------
__global__ __launch_bounds__(256)
void transpose_f32_to_f16_kernel(const float* __restrict__ in, _Float16* __restrict__ out,
                                 int K, int N)
{
    __shared__ float tile[32][33];
    int bx = blockIdx.x * 32;       // N offset
    int by = blockIdx.y * 32;       // K offset
    int tx = threadIdx.x & 31;
    int ty = threadIdx.x >> 5;      // 0..7
#pragma unroll
    for (int yy = ty; yy < 32; yy += 8)
        tile[yy][tx] = in[(size_t)(by + yy) * N + (bx + tx)];
    __syncthreads();
#pragma unroll
    for (int yy = ty; yy < 32; yy += 8)
        out[(size_t)(bx + yy) * K + (by + tx)] = (_Float16)tile[tx][yy];
}

// ---------------------------------------------------------------------------
// 3) WMMA GEMM: C[M,N] = A[M,K]f16 @ BT[N,K]f16^T + bias, optional ReLU.
//    One 16x16 C tile per wave; fragments per CDNA5 ISA 7.12.2 layouts.
//    A and BT are both K-contiguous -> all fragment loads are b128.
// ---------------------------------------------------------------------------
#define GEMM_WAVES 4
template<bool RELU, bool STORE_F, bool STORE_H>
__global__ __launch_bounds__(128)
void gemm_wmma_f16(const _Float16* __restrict__ A, const _Float16* __restrict__ BT,
                   const float* __restrict__ bias,
                   float* __restrict__ outF, _Float16* __restrict__ outH,
                   int M, int N, int Kd)
{
    int lane = threadIdx.x & 31;
    int wave = threadIdx.x >> 5;
    int g    = lane >> 4;        // lane group (0: lanes 0-15, 1: lanes 16-31)
    int r    = lane & 15;
    int row0 = blockIdx.y * 16;
    int col0 = (blockIdx.x * GEMM_WAVES + wave) * 16;

    v8f acc = {};
    // A fragment rows: lane-group g holds K = {8g+0..7} and {16+8g+0..7}
    const _Float16* Arow = A  + (size_t)(row0 + r) * Kd + 8 * g;    // 16B-aligned
    // B fragment cols: lane-group g holds K = {8g+0..7} and {16+8g+0..7}
    const _Float16* Brow = BT + (size_t)(col0 + r) * Kd + 8 * g;    // 16B-aligned

    for (int k0 = 0; k0 < Kd; k0 += 32) {
        v8h alo = *(const v8h*)(Arow + k0);        // global_load_b128
        v8h ahi = *(const v8h*)(Arow + k0 + 16);   // global_load_b128
        v8h blo = *(const v8h*)(Brow + k0);        // global_load_b128
        v8h bhi = *(const v8h*)(Brow + k0 + 16);   // global_load_b128
        v16h a, b;
#pragma unroll
        for (int e = 0; e < 8; ++e) {
            a[e] = alo[e]; a[e + 8] = ahi[e];
            b[e] = blo[e]; b[e + 8] = bhi[e];
        }
        __builtin_prefetch(Arow + k0 + 64, 0, 1);  // global_prefetch_b8 next block
        acc = __builtin_amdgcn_wmma_f32_16x16x32_f16(
            /*neg_a=*/false, a, /*neg_b=*/false, b,
            /*c_mod=*/(short)0, acc, /*reuse_a=*/false, /*reuse_b=*/false);
    }

    int col = col0 + r;
    float bval = bias[col];
#pragma unroll
    for (int v = 0; v < 8; ++v) {
        int row = row0 + v + 8 * g;               // C/D layout: VGPR v, group g
        float x = acc[v] + bval;
        if (RELU) x = fmaxf(x, 0.0f);
        size_t off = (size_t)row * N + col;
        if (STORE_F) outF[off] = x;
        if (STORE_H) outH[off] = (_Float16)x;
    }
}

// ---------------------------------------------------------------------------
// 4) Temporal MLP: t = relu(ts*Wt1 + bt1) @ Wt2 + bt2   ([S,128] -> [S,256])
// ---------------------------------------------------------------------------
__global__ __launch_bounds__(256)
void temporal_kernel(const float* __restrict__ ts_s, const float* __restrict__ Wt1,
                     const float* __restrict__ bt1, const float* __restrict__ Wt2,
                     const float* __restrict__ bt2, float* __restrict__ t_out)
{
    __shared__ float t1[128];
    int row = blockIdx.x;
    int tid = threadIdx.x;
    float tv = ts_s[row];
    if (tid < 128) t1[tid] = fmaxf(tv * Wt1[tid] + bt1[tid], 0.0f);
    __syncthreads();
    float acc = bt2[tid];
    for (int k = 0; k < 128; ++k) acc += t1[k] * Wt2[k * 256 + tid];
    t_out[row * 256 + tid] = acc;
}

// ---------------------------------------------------------------------------
// 5) LayerNorm over H=512; optionally fuse sinusoidal positional encoding.
// ---------------------------------------------------------------------------
template<bool STORE_F, bool STORE_H, bool ADD_TE>
__global__ __launch_bounds__(256)
void layernorm_kernel(const float* __restrict__ in, const float* __restrict__ gamma,
                      const float* __restrict__ beta, float* __restrict__ outF,
                      _Float16* __restrict__ outH)
{
    __shared__ float red[256];
    int row = blockIdx.x;
    int tid = threadIdx.x;
    const float* x = in + (size_t)row * H;
    float a = x[tid], b = x[tid + 256];
    red[tid] = a + b;
    __syncthreads();
    for (int s = 128; s > 0; s >>= 1) { if (tid < s) red[tid] += red[tid + s]; __syncthreads(); }
    float mean = red[0] * (1.0f / H);
    __syncthreads();
    float da = a - mean, db = b - mean;
    red[tid] = da * da + db * db;
    __syncthreads();
    for (int s = 128; s > 0; s >>= 1) { if (tid < s) red[tid] += red[tid + s]; __syncthreads(); }
    float rstd = rsqrtf(red[0] * (1.0f / H) + 1e-5f);

#pragma unroll
    for (int half = 0; half < 2; ++half) {
        int c = tid + half * 256;
        float val = (half ? db : da) * rstd * gamma[c] + beta[c];
        if (ADD_TE) {
            int k = c >> 1;
            float div = __expf((float)(2 * k) * (-9.210340371976184f / (float)H));
            float ang = (float)row * div;
            val += (c & 1) ? __cosf(ang) : __sinf(ang);
        }
        size_t off = (size_t)row * H + c;
        if (STORE_F) outF[off] = val;
        if (STORE_H) outH[off] = (_Float16)val;
    }
}

// ---------------------------------------------------------------------------
// 6) comb = amps * concat(h, tile(t,2))  -> f16 [S, 2H]
// ---------------------------------------------------------------------------
__global__ __launch_bounds__(256)
void build_comb_kernel(const float* __restrict__ h, const float* __restrict__ t,
                       const float* __restrict__ amps_s, _Float16* __restrict__ comb)
{
    int row = blockIdx.x;
    float amp = amps_s[row];
    for (int c = threadIdx.x; c < 2 * H; c += 256) {
        float v = (c < H) ? h[(size_t)row * H + c]
                          : t[(size_t)row * 256 + ((c - H) & 255)];
        comb[(size_t)row * (2 * H) + c] = (_Float16)(v * amp);
    }
}

// ---------------------------------------------------------------------------
// 7) K fan-out: K[i][j][:] = Kbase[j][:] + 0.1/(1+|i-j|).  512 MiB streaming
//    writes -> nontemporal b128 stores (bypass L2 dirty fill).
//    Block (j, i-chunk): Kbase row in registers, 64 i-rows per block.
// ---------------------------------------------------------------------------
#define FAN_ICHUNK 64
__global__ __launch_bounds__(128)
void k_fanout_kernel(const float* __restrict__ Kbase, float* __restrict__ outK)
{
    int j   = blockIdx.x;
    int i0  = blockIdx.y * FAN_ICHUNK;
    int tid = threadIdx.x;                 // 128 threads x v4f = 512 floats
    v4f base = ((const v4f*)(Kbase + (size_t)j * H))[tid];
    for (int i = i0; i < i0 + FAN_ICHUNK; ++i) {
        int d = i - j; if (d < 0) d = -d;
        float c = 0.1f / (1.0f + (float)d);
        v4f v = base + c;                  // scalar broadcast add
        v4f* dst = (v4f*)(outK + ((size_t)i * S + j) * H) + tid;
        __builtin_nontemporal_store(v, dst);
    }
}

// ---------------------------------------------------------------------------
// Host orchestration
// ---------------------------------------------------------------------------
extern "C" void kernel_launch(void* const* d_in, const int* in_sizes, int n_in,
                              void* d_out, int out_size, void* d_ws, size_t ws_size,
                              hipStream_t stream)
{
    const float* ts    = (const float*)d_in[0];
    const float* feats = (const float*)d_in[1];
    const float* amps  = (const float*)d_in[2];
    // d_in[3] neuron_ids: unused by reference math
    const float* We1 = (const float*)d_in[4];  const float* be1 = (const float*)d_in[5];
    const float* We2 = (const float*)d_in[6];  const float* be2 = (const float*)d_in[7];
    const float* gev = (const float*)d_in[8];  const float* bev = (const float*)d_in[9];
    const float* Wt1 = (const float*)d_in[10]; const float* bt1 = (const float*)d_in[11];
    const float* Wt2 = (const float*)d_in[12]; const float* bt2 = (const float*)d_in[13];
    const float* Wc  = (const float*)d_in[14]; const float* bc  = (const float*)d_in[15];
    const float* gsr = (const float*)d_in[16]; const float* bsr = (const float*)d_in[17];
    const float* Wq  = (const float*)d_in[18]; const float* bq  = (const float*)d_in[19];
    const float* Wk  = (const float*)d_in[20]; const float* bk  = (const float*)d_in[21];
    const float* Wv  = (const float*)d_in[22]; const float* bv  = (const float*)d_in[23];

    // Bump allocator over the workspace (all offsets 256B aligned).
    char* p = (char*)d_ws;
    auto alloc = [&](size_t bytes) -> char* {
        char* r = p; p += (bytes + 255) & ~(size_t)255; return r;
    };
    int*      order   = (int*)     alloc(S * sizeof(int));
    float*    ts_s    = (float*)   alloc(S * sizeof(float));
    float*    amps_s  = (float*)   alloc(S * sizeof(float));
    _Float16* featsH  = (_Float16*)alloc((size_t)S * DIN * 2);
    _Float16* We1t    = (_Float16*)alloc((size_t)DIN * 256 * 2);   // [256, DIN]
    _Float16* We2t    = (_Float16*)alloc((size_t)256 * H * 2);     // [H, 256]
    _Float16* Wct     = (_Float16*)alloc((size_t)(2 * H) * H * 2); // [H, 2H]
    _Float16* Wqt     = (_Float16*)alloc((size_t)H * H * 2);       // [H, H]
    _Float16* Wkt     = (_Float16*)alloc((size_t)H * H * 2);
    _Float16* Wvt     = (_Float16*)alloc((size_t)H * H * 2);
    _Float16* h1H     = (_Float16*)alloc((size_t)S * 256 * 2);
    float*    hpre    = (float*)   alloc((size_t)S * H * 4);
    float*    hF      = (float*)   alloc((size_t)S * H * 4);
    float*    tF      = (float*)   alloc((size_t)S * 256 * 4);
    _Float16* combH   = (_Float16*)alloc((size_t)S * 2 * H * 2);
    float*    combpre = (float*)   alloc((size_t)S * H * 4);
    _Float16* ctxH    = (_Float16*)alloc((size_t)S * H * 2);
    float*    Kbase   = (float*)   alloc((size_t)S * H * 4);

    float* outQ = (float*)d_out;                               // [1,S,H]
    float* outK = outQ + (size_t)S * H;                        // [S,S,H]
    float* outV = outK + (size_t)S * S * H;                    // [1,S,H]

    // Sort + gathers
    sort_rank_kernel<<<1, 512, 0, stream>>>(ts, amps, order, ts_s, amps_s);
    gather_feats_kernel<<<S, 256, 0, stream>>>(feats, order, featsH);

    // Weight transpose-conversions: W[K,N] f32 -> WT[N,K] f16
    auto cvtT = [&](const float* src, _Float16* dst, int K, int N) {
        transpose_f32_to_f16_kernel<<<dim3(N / 32, K / 32), 256, 0, stream>>>(src, dst, K, N);
    };
    cvtT(We1, We1t, DIN, 256);
    cvtT(We2, We2t, 256, H);
    cvtT(Wc,  Wct,  2 * H, H);
    cvtT(Wq,  Wqt,  H, H);
    cvtT(Wk,  Wkt,  H, H);
    cvtT(Wv,  Wvt,  H, H);

    // Event encoder MLP:  h1 = relu(feats@We1+be1) ; hpre = h1@We2+be2
    gemm_wmma_f16<true, false, true><<<dim3(256 / (16 * GEMM_WAVES), S / 16), 128, 0, stream>>>(
        featsH, We1t, be1, nullptr, h1H, S, 256, DIN);
    gemm_wmma_f16<false, true, false><<<dim3(H / (16 * GEMM_WAVES), S / 16), 128, 0, stream>>>(
        h1H, We2t, be2, hpre, nullptr, S, H, 256);
    layernorm_kernel<true, false, false><<<S, 256, 0, stream>>>(hpre, gev, bev, hF, nullptr);

    // Temporal encoder + combine
    temporal_kernel<<<S, 256, 0, stream>>>(ts_s, Wt1, bt1, Wt2, bt2, tF);
    build_comb_kernel<<<S, 256, 0, stream>>>(hF, tF, amps_s, combH);

    // comb @ Wc + bc -> LN (+ sinusoidal PE) -> ctx (f16)
    gemm_wmma_f16<false, true, false><<<dim3(H / (16 * GEMM_WAVES), S / 16), 128, 0, stream>>>(
        combH, Wct, bc, combpre, nullptr, S, H, 2 * H);
    layernorm_kernel<false, true, true><<<S, 256, 0, stream>>>(combpre, gsr, bsr, nullptr, ctxH);

    // Q / Kbase / V projections
    gemm_wmma_f16<false, true, false><<<dim3(H / (16 * GEMM_WAVES), S / 16), 128, 0, stream>>>(
        ctxH, Wqt, bq, outQ, nullptr, S, H, H);
    gemm_wmma_f16<false, true, false><<<dim3(H / (16 * GEMM_WAVES), S / 16), 128, 0, stream>>>(
        ctxH, Wkt, bk, Kbase, nullptr, S, H, H);
    gemm_wmma_f16<false, true, false><<<dim3(H / (16 * GEMM_WAVES), S / 16), 128, 0, stream>>>(
        ctxH, Wvt, bv, outV, nullptr, S, H, H);

    // K broadcast fan-out: the 512 MiB streaming-store stage.
    k_fanout_kernel<<<dim3(S, S / FAN_ICHUNK), 128, 0, stream>>>(Kbase, outK);
}
